// EchoStateNetwork_63728724738389
// MI455X (gfx1250) — matrix operand order repaired and checked by reading, not compiled, not run on hardware
//
#include <hip/hip_runtime.h>
#include <hip/hip_bf16.h>

// ---------------- problem constants ----------------
#define B_    256
#define T_    512
#define HID_  1024
#define DISC_ 64
#define TK_   448                      // kept timesteps (T - DISC)
#define HOUT_ 24
#define ROSZ_ ((1 + HID_) * TK_)       // 459200

typedef __attribute__((ext_vector_type(16))) __bf16 v16bf;
typedef __attribute__((ext_vector_type(8)))  __bf16 v8bf;
typedef __attribute__((ext_vector_type(8)))  float  v8f;

#define SHUF16(lo, hi) __builtin_shufflevector(lo, hi, 0,1,2,3,4,5,6,7,8,9,10,11,12,13,14,15)

// bf16 h mirror row stride: 1032 elems = 2064 B (16B aligned; 516 dw == 4 mod 64 banks)
#define HB_STRIDE 1032
#define HB_BYTES  (16 * HB_STRIDE * 2)          // 33,024 B
#define RRED_OFF  HB_BYTES                      // 32 waves * 512 f32 = 65,536 B
#define XSH_OFF   (RRED_OFF + 32 * 512 * 4)
#define SMEM_BYTES (XSH_OFF + 2 * 16 * 4)

// fast tanh from hardware transcendentals:
// tanh(v) = (e^{2v}-1)/(e^{2v}+1) = 1 - 2/(e^{2v}+1),  e^{2v} = exp2(2*log2(e)*v)
__device__ __forceinline__ float fast_tanhf(float v)
{
    float e = __builtin_amdgcn_exp2f(v * 2.8853900817779268f); // 2*log2(e)
    return 1.0f - 2.0f * __builtin_amdgcn_rcpf(e + 1.0f);
}

// ---------------------------------------------------------------------------
// Pack W_hh (f32 [HID][HID], row-major) into bf16 B-fragments for
// v_wmma_f32_16x16x32_bf16:  Bmat[k][n] = W_hh[n][k]  (since C = h @ W_hh^T).
// Fragment block = (ntile, kchunk): 32x16 bf16, 1 KB; lane l holds 32 contiguous
// bytes = K = kchunk*32 + (l>>4)*8 + [0..7], then +16, col n = ntile*16 + (l&15).
// 2048 blocks -> 2 MB (stays resident in the 192 MB L2).
// ---------------------------------------------------------------------------
__global__ void esn_pack_whh(const float* __restrict__ whh, __bf16* __restrict__ wpack)
{
    int e    = blockIdx.x * 256 + threadIdx.x;   // 1,048,576 elements total
    int j2   = e & 15;
    int lane = (e >> 4) & 31;
    int blk  = e >> 9;                           // ntile*32 + kchunk
    int ntile = blk >> 5;
    int kc    = blk & 31;
    int n = ntile * 16 + (lane & 15);
    int k = kc * 32 + ((lane >> 4) * 8) + (j2 & 7) + ((j2 >> 3) * 16);
    wpack[e] = (__bf16)whh[n * HID_ + k];
}

// ---------------------------------------------------------------------------
// Pack the hidden part of W_ro into per-(timestep, wave) bf16 B-fragments for
// the fused readout WMMA.  For kept step tt and wave w (owning hid columns
// ng = w*32 + nl), tile o-range [tile*16, tile*16+16) (cols 24..31 zero):
//   value = W_ro[o * ROSZ_ + 448 + ng*448 + tt]
// 448 * 32 * 2 blocks * 1 KB = 28 MB (L2-resident).
// ---------------------------------------------------------------------------
__global__ void esn_pack_wro(const float* __restrict__ wro, __bf16* __restrict__ wrof)
{
    size_t e  = (size_t)blockIdx.x * 256 + threadIdx.x;  // 14,680,064 total
    int j2    = (int)(e & 15);
    int lane  = (int)((e >> 4) & 31);
    int tile  = (int)((e >> 9) & 1);
    size_t e2 = e >> 10;
    int wave  = (int)(e2 & 31);
    int tt    = (int)(e2 >> 5);
    int o  = tile * 16 + (lane & 15);
    int nl = (lane >> 4) * 8 + (j2 & 7) + ((j2 >> 3) * 16);
    int ng = wave * 32 + nl;
    float v = 0.0f;
    if (o < HOUT_) v = wro[(size_t)o * ROSZ_ + TK_ + (size_t)ng * TK_ + tt];
    wrof[e] = (__bf16)v;
}

// ---------------------------------------------------------------------------
// Main kernel: 16 workgroups (one per 16-row batch tile), 1024 threads = 32
// waves.  Wave w owns output/hid columns [w*32, w*32+32) = two 16x16 N-tiles.
// f32 carried state h lives in VGPRs (C-tile layout: m = r + 8*(lane>>4),
// n = ntile*16 + (lane&15)); a bf16 mirror in LDS feeds cross-wave A-frags.
// Per step: 64 main WMMAs + 2 fused-readout WMMAs per wave, two barriers.
// ---------------------------------------------------------------------------
__global__ __launch_bounds__(1024, 1)
void esn_main(const float* __restrict__ x,       // [B*T]
              const __bf16* __restrict__ wpack,  // packed W_hh^T fragments
              const float* __restrict__ win,     // [HID] (W_in, D=1)
              const float* __restrict__ bin,     // [HID]
              const __bf16* __restrict__ wrof,   // packed readout fragments
              const float* __restrict__ wro,     // raw W_ro (x_io part)
              const float* __restrict__ bro,     // [HOUT]
              float* __restrict__ out)           // [B*HOUT]
{
    extern __shared__ char smem[];
    __bf16* hb   = (__bf16*)smem;                   // bf16 mirror of h [16][HB_STRIDE]
    float*  rred = (float*)(smem + RRED_OFF);       // cross-wave readout partials
    float*  xsh  = (float*)(smem + XSH_OFF);        // x staging, double buffered [2][16]

    const int tid  = threadIdx.x;
    const int wave = tid >> 5;
    const int lane = tid & 31;
    const int b0   = blockIdx.x * 16;

    // ---- init: h = 0 ----
    for (int i = tid; i < 16 * HB_STRIDE; i += 1024) hb[i] = (__bf16)0.0f;
    if (tid < 16) xsh[tid] = x[(size_t)(b0 + tid) * T_];   // buffer 0, t = 0
    __syncthreads();

    const int ncol  = lane & 15;
    const int hi    = lane >> 4;
    const int akoff = hi * 8;                  // A-frag: dwords 0-3 K=hi*8.., 4-7 K=16+hi*8..
    const int nt0 = wave * 2, nt1 = nt0 + 1;
    const float wi0 = win[nt0 * 16 + ncol], bi0 = bin[nt0 * 16 + ncol];
    const float wi1 = win[nt1 * 16 + ncol], bi1 = bin[nt1 * 16 + ncol];

    const __bf16* wbase0 = wpack + (size_t)(nt0 * 32) * 512 + lane * 16;
    const __bf16* wbase1 = wpack + (size_t)(nt1 * 32) * 512 + lane * 16;
    const __bf16* abase  = hb + ncol * HB_STRIDE + akoff;   // A-frag row = lane&15
    const __bf16* a2base = abase + wave * 32;               // readout A: K = own n-range

    v8f hold0 = {}, hold1 = {};      // f32 carried state (this wave's columns)
    v8f racc0 = {}, racc1 = {};      // fused readout accumulators (persist over t)

    for (int t = 0; t < T_; ++t) {
        // ---- phase 1: C = h_{t-1} @ W_hh^T  (reads hb only) ----
        v8f c0 = {}, c1 = {};
        #pragma unroll 4
        for (int kc = 0; kc < 32; ++kc) {
            v8bf alo = *(const v8bf*)(abase + kc * 32);
            v8bf ahi = *(const v8bf*)(abase + kc * 32 + 16);
            v16bf a  = SHUF16(alo, ahi);
            v16bf bb0 = *(const v16bf*)(wbase0 + kc * 512);
            v16bf bb1 = *(const v16bf*)(wbase1 + kc * 512);
            c0 = __builtin_amdgcn_wmma_f32_16x16x32_bf16(false, a, false, bb0, (short)0, c0, false, false);
            c1 = __builtin_amdgcn_wmma_f32_16x16x32_bf16(false, a, false, bb1, (short)0, c1, false, false);
        }
        // ---- fused readout on h_{t-1} (kept state index s = t-1 >= DISC) ----
        if (t >= DISC_ + 1) {
            int tt = t - (DISC_ + 1);
            v8bf alo = *(const v8bf*)(a2base);
            v8bf ahi = *(const v8bf*)(a2base + 16);
            v16bf a2 = SHUF16(alo, ahi);
            const __bf16* rb = wrof + (size_t)((tt * 32 + wave) * 2) * 512 + lane * 16;
            v16bf rb0 = *(const v16bf*)rb;
            v16bf rb1 = *(const v16bf*)(rb + 512);
            racc0 = __builtin_amdgcn_wmma_f32_16x16x32_bf16(false, a2, false, rb0, (short)0, racc0, false, false);
            racc1 = __builtin_amdgcn_wmma_f32_16x16x32_bf16(false, a2, false, rb1, (short)0, racc1, false, false);
        }
        __syncthreads();   // all hb reads done before updating h

        // ---- phase 2: h_t = 0.5*h_{t-1} + 0.5*tanh(C + x_t*w_in + b_in) ----
        const int xbuf = (t & 1) * 16;
        #pragma unroll
        for (int r = 0; r < 8; ++r) {
            int m = r + 8 * hi;
            float xv = xsh[xbuf + m];
            float hn0 = 0.5f * (hold0[r] + fast_tanhf(c0[r] + xv * wi0 + bi0));
            float hn1 = 0.5f * (hold1[r] + fast_tanhf(c1[r] + xv * wi1 + bi1));
            hold0[r] = hn0;
            hold1[r] = hn1;
            hb[m * HB_STRIDE + nt0 * 16 + ncol] = (__bf16)hn0;
            hb[m * HB_STRIDE + nt1 * 16 + ncol] = (__bf16)hn1;
        }
        if (tid < 16 && t + 1 < T_)
            xsh[((t + 1) & 1) * 16 + tid] = x[(size_t)(b0 + tid) * T_ + t + 1];
        __syncthreads();   // hb writes visible before next step's reads
    }

    // ---- tail readout for the final state s = T-1 (tt = 447) ----
    {
        int tt = TK_ - 1;
        v8bf alo = *(const v8bf*)(a2base);
        v8bf ahi = *(const v8bf*)(a2base + 16);
        v16bf a2 = SHUF16(alo, ahi);
        const __bf16* rb = wrof + (size_t)((tt * 32 + wave) * 2) * 512 + lane * 16;
        v16bf rb0 = *(const v16bf*)rb;
        v16bf rb1 = *(const v16bf*)(rb + 512);
        racc0 = __builtin_amdgcn_wmma_f32_16x16x32_bf16(false, a2, false, rb0, (short)0, racc0, false, false);
        racc1 = __builtin_amdgcn_wmma_f32_16x16x32_bf16(false, a2, false, rb1, (short)0, racc1, false, false);
    }

    // ---- cross-wave reduction of readout partials (waves are K-slices) ----
    #pragma unroll
    for (int r = 0; r < 8; ++r) {
        int m = r + 8 * hi;
        rred[wave * 512 + m * 32 + ncol]      = racc0[r];
        rred[wave * 512 + m * 32 + 16 + ncol] = racc1[r];
    }
    __syncthreads();

    if (tid < 16 * HOUT_) {
        int m = tid / HOUT_, o = tid % HOUT_;
        float acc = bro[o];
        #pragma unroll 8
        for (int w2 = 0; w2 < 32; ++w2) acc += rred[w2 * 512 + m * 32 + o];
        // x_io contribution: feat[0..447] = x[b, DISC:]
        const float* wr = wro + (size_t)o * ROSZ_;
        const float* xr = x + (size_t)(b0 + m) * T_ + DISC_;
        #pragma unroll 4
        for (int tt2 = 0; tt2 < TK_; ++tt2) acc += xr[tt2] * wr[tt2];
        out[(b0 + m) * HOUT_ + o] = acc;
    }
}

// ---------------------------------------------------------------------------
extern "C" void kernel_launch(void* const* d_in, const int* in_sizes, int n_in,
                              void* d_out, int out_size, void* d_ws, size_t ws_size,
                              hipStream_t stream)
{
    const float* x    = (const float*)d_in[0];   // [B,1,T]
    const float* W_in = (const float*)d_in[1];   // [HID,1]
    const float* b_in = (const float*)d_in[2];   // [HID]
    const float* W_hh = (const float*)d_in[3];   // [HID,HID]
    const float* W_ro = (const float*)d_in[4];   // [HOUT,ROSZ]
    const float* b_ro = (const float*)d_in[5];   // [HOUT]
    float* out = (float*)d_out;

    __bf16* wpack = (__bf16*)d_ws;                                    // 2 MB
    __bf16* wrof  = (__bf16*)((char*)d_ws + (size_t)2 * 1024 * 1024); // 28 MB

    esn_pack_whh<<<4096, 256, 0, stream>>>(W_hh, wpack);
    esn_pack_wro<<<57344, 256, 0, stream>>>(W_ro, wrof);
    esn_main<<<16, 1024, SMEM_BYTES, stream>>>(x, wpack, W_in, b_in, wrof, W_ro, b_ro, out);
}
